// AdvancedLiquidNeuralNetwork_26766236188867
// MI455X (gfx1250) — compile-verified, matrix operand order checked by reading, not run on hardware
//
#include <hip/hip_runtime.h>
#include <math.h>

typedef __attribute__((ext_vector_type(16))) __bf16 v16bf;
typedef __attribute__((ext_vector_type(8)))  float  v8f;
typedef __attribute__((ext_vector_type(8)))  int    v8i;

#define HDIM 512
#define INDIM 256
#define OUTDIM 128
#define MB 16            // rows per workgroup (head kernel)
#define MB2 32           // rows per workgroup (step / xin kernels)
#define NB 2048          // 32768 / MB
#define NB2 1024         // 32768 / MB2
#define ASTR 520         // bf16 LDS stride (1040B, 16B aligned, bank-skewed)

#define Z8 {0.f,0.f,0.f,0.f,0.f,0.f,0.f,0.f}

__device__ __forceinline__ unsigned short f2bf(float f) {
  unsigned u = __builtin_bit_cast(unsigned, f);
  u += 0x7FFFu + ((u >> 16) & 1u);          // round-to-nearest-even
  return (unsigned short)(u >> 16);
}

// Load one 16x32 bf16 operand fragment (A or B) for v_wmma_f32_16x16x32_bf16.
// Per ISA 7.12.2: lane%16 selects the row; VGPRs 0-3 hold K = kc + 8*(lane/16) .. +7,
// VGPRs 4-7 hold K = kc + 16 + 8*(lane/16) .. +7  -> two contiguous b128 loads.
__device__ __forceinline__ v16bf load_frag(const unsigned short* p) {
  int4 lo = *(const int4*)(p);
  int4 hi = *(const int4*)(p + 16);
  v8i t;
  t[0] = lo.x; t[1] = lo.y; t[2] = lo.z; t[3] = lo.w;
  t[4] = hi.x; t[5] = hi.y; t[6] = hi.z; t[7] = hi.w;
  return __builtin_bit_cast(v16bf, t);
}

__device__ __forceinline__ v8f wmma_bf16(v16bf a, v16bf b, v8f c) {
  return __builtin_amdgcn_wmma_f32_16x16x32_bf16(false, a, false, b, (short)0, c, false, false);
}

__global__ void __launch_bounds__(256) lnn_cvt_bf16(const float* __restrict__ s,
                                                    unsigned short* __restrict__ d, int n) {
  int i = blockIdx.x * 256 + threadIdx.x;
  if (i < n) d[i] = f2bf(s[i]);
}

__global__ void __launch_bounds__(256) lnn_zero_f32(float* __restrict__ p, int n) {
  int i = blockIdx.x * 256 + threadIdx.x;
  if (i < n) p[i] = 0.f;
}

// xin = x @ W_in^T + b_in    [32768,256] x [512,256]^T -> [32768,512]
__global__ void __launch_bounds__(256) lnn_xin_kernel(
    const float* __restrict__ x, const unsigned short* __restrict__ wIn,
    const float* __restrict__ bIn, float* __restrict__ xin) {
  __shared__ __align__(16) unsigned short xL[MB2 * 264];
  const int tid = threadIdx.x;
  const int row0 = blockIdx.x * MB2;
  for (int i = tid; i < MB2 * INDIM; i += 256) {
    int r = i >> 8, c = i & 255;
    xL[r * 264 + c] = f2bf(x[(size_t)(row0 + r) * INDIM + c]);
  }
  __syncthreads();
  const int lane = tid & 31, wave = tid >> 5;
  const int hi = lane >> 4, ls = lane & 15;
  const int n0 = wave * 64;
  v8f acc[2][4] = {{Z8, Z8, Z8, Z8}, {Z8, Z8, Z8, Z8}};
  for (int kc = 0; kc < INDIM; kc += 32) {
    v16bf a0 = load_frag(&xL[ls * 264 + kc + hi * 8]);
    v16bf a1 = load_frag(&xL[(16 + ls) * 264 + kc + hi * 8]);
#pragma unroll
    for (int nt = 0; nt < 4; ++nt) {
      v16bf b = load_frag(wIn + (size_t)(n0 + nt * 16 + ls) * INDIM + kc + hi * 8);
      acc[0][nt] = wmma_bf16(a0, b, acc[0][nt]);
      acc[1][nt] = wmma_bf16(a1, b, acc[1][nt]);
    }
  }
#pragma unroll
  for (int mt = 0; mt < 2; ++mt)
#pragma unroll
    for (int nt = 0; nt < 4; ++nt) {
      int col = n0 + nt * 16 + ls;
      float bb = bIn[col];
#pragma unroll
      for (int r = 0; r < 8; ++r) {
        int m = mt * 16 + r + hi * 8;
        xin[(size_t)(row0 + m) * HDIM + col] = acc[mt][nt][r] + bb;
      }
    }
}

// One full liquid time step, fused: 3x (GEMM + LN) summed, W_v GEMM, W_o GEMM,
// LN, tanh state update. Each block owns 32 rows x all 512 cols.
// LN row stats are computed with half-wave shfl_xor butterflies (the 16x16 C/D
// layout keeps each row inside one 16-lane half), then 8 wave partials are summed via f32 LDS.
__global__ void __launch_bounds__(256) lnn_step_kernel(
    float* __restrict__ h, const float* __restrict__ xin,
    const unsigned short* __restrict__ wLiq, const float* __restrict__ bLiq,
    const float* __restrict__ gLiq, const float* __restrict__ betaLiq,
    const unsigned short* __restrict__ wV, const float* __restrict__ bV,
    const unsigned short* __restrict__ wO, const float* __restrict__ bO,
    const float* __restrict__ gAtt, const float* __restrict__ betaAtt,
    const float* __restrict__ tau, unsigned short* __restrict__ h0bf, int write_h0) {
  __shared__ __align__(16) unsigned short Abuf[MB2 * ASTR];  // 33,280 B
  __shared__ float rowS[8][MB2];
  __shared__ float rowQ[8][MB2];
  __shared__ float meanA[MB2], rstdA[MB2];

  const int tid = threadIdx.x;
  const int row0 = blockIdx.x * MB2;
  const int lane = tid & 31, wave = tid >> 5;
  const int hi = lane >> 4, ls = lane & 15;
  const int n0 = wave * 64;

  // stage h (bf16) as the shared A operand
  for (int i = tid; i < MB2 * HDIM; i += 256) {
    int r = i >> 9, c = i & 511;
    Abuf[r * ASTR + c] = f2bf(h[(size_t)(row0 + r) * HDIM + c]);
  }
  __syncthreads();

  v8f liq[2][4] = {{Z8, Z8, Z8, Z8}, {Z8, Z8, Z8, Z8}};
  v8f acc[2][4];

  // ---- 3 liquid layers: pre_l = h + h@W_l^T + b_l ; liq += LN(pre_l) ----
  for (int l = 0; l < 3; ++l) {
    const unsigned short* W = wLiq + (size_t)l * HDIM * HDIM;
#pragma unroll
    for (int mt = 0; mt < 2; ++mt)
#pragma unroll
      for (int nt = 0; nt < 4; ++nt) acc[mt][nt] = (v8f)Z8;
    for (int kc = 0; kc < HDIM; kc += 32) {
      v16bf a0 = load_frag(&Abuf[ls * ASTR + kc + hi * 8]);
      v16bf a1 = load_frag(&Abuf[(16 + ls) * ASTR + kc + hi * 8]);
#pragma unroll
      for (int nt = 0; nt < 4; ++nt) {
        v16bf b = load_frag(W + (size_t)(n0 + nt * 16 + ls) * HDIM + kc + hi * 8);
        acc[0][nt] = wmma_bf16(a0, b, acc[0][nt]);
        acc[1][nt] = wmma_bf16(a1, b, acc[1][nt]);
      }
    }
    const float* bl = bLiq + l * HDIM;
    const float* gl = gLiq + l * HDIM;
    const float* btl = betaLiq + l * HDIM;
    // pre = gemm + h + bias
#pragma unroll
    for (int mt = 0; mt < 2; ++mt)
#pragma unroll
      for (int nt = 0; nt < 4; ++nt) {
        int col = n0 + nt * 16 + ls;
        float bb = bl[col];
#pragma unroll
        for (int r = 0; r < 8; ++r) {
          int m = mt * 16 + r + hi * 8;
          acc[mt][nt][r] += h[(size_t)(row0 + m) * HDIM + col] + bb;
        }
      }
    // row stats: butterfly over the 16-lane half, then combine 8 wave partials
#pragma unroll
    for (int mt = 0; mt < 2; ++mt)
#pragma unroll
      for (int r = 0; r < 8; ++r) {
        float s = acc[mt][0][r] + acc[mt][1][r] + acc[mt][2][r] + acc[mt][3][r];
        float q = acc[mt][0][r] * acc[mt][0][r] + acc[mt][1][r] * acc[mt][1][r] +
                  acc[mt][2][r] * acc[mt][2][r] + acc[mt][3][r] * acc[mt][3][r];
#pragma unroll
        for (int msk = 1; msk < 16; msk <<= 1) {
          s += __shfl_xor(s, msk, 32);
          q += __shfl_xor(q, msk, 32);
        }
        if (ls == 0) {
          int m = mt * 16 + r + hi * 8;
          rowS[wave][m] = s;
          rowQ[wave][m] = q;
        }
      }
    __syncthreads();
    if (tid < MB2) {
      float s = 0.f, q = 0.f;
#pragma unroll
      for (int w = 0; w < 8; ++w) { s += rowS[w][tid]; q += rowQ[w][tid]; }
      float mean = s * (1.f / 512.f);
      float var = q * (1.f / 512.f) - mean * mean;
      meanA[tid] = mean;
      rstdA[tid] = rsqrtf(var + 1e-5f);
    }
    __syncthreads();
#pragma unroll
    for (int mt = 0; mt < 2; ++mt)
#pragma unroll
      for (int nt = 0; nt < 4; ++nt) {
        int col = n0 + nt * 16 + ls;
        float g = gl[col], bt = btl[col];
#pragma unroll
        for (int r = 0; r < 8; ++r) {
          int m = mt * 16 + r + hi * 8;
          liq[mt][nt][r] += (acc[mt][nt][r] - meanA[m]) * rstdA[m] * g + bt;
        }
      }
    __syncthreads();
  }

  // ---- GEMM2: tmp = liq @ W_v^T + b_v (liq bf16 through LDS) ----
#pragma unroll
  for (int mt = 0; mt < 2; ++mt)
#pragma unroll
    for (int nt = 0; nt < 4; ++nt) {
      int col = n0 + nt * 16 + ls;
#pragma unroll
      for (int r = 0; r < 8; ++r)
        Abuf[(mt * 16 + r + hi * 8) * ASTR + col] = f2bf(liq[mt][nt][r]);
    }
  __syncthreads();
#pragma unroll
  for (int mt = 0; mt < 2; ++mt)
#pragma unroll
    for (int nt = 0; nt < 4; ++nt) acc[mt][nt] = (v8f)Z8;
  for (int kc = 0; kc < HDIM; kc += 32) {
    v16bf a0 = load_frag(&Abuf[ls * ASTR + kc + hi * 8]);
    v16bf a1 = load_frag(&Abuf[(16 + ls) * ASTR + kc + hi * 8]);
#pragma unroll
    for (int nt = 0; nt < 4; ++nt) {
      v16bf b = load_frag(wV + (size_t)(n0 + nt * 16 + ls) * HDIM + kc + hi * 8);
      acc[0][nt] = wmma_bf16(a0, b, acc[0][nt]);
      acc[1][nt] = wmma_bf16(a1, b, acc[1][nt]);
    }
  }
  __syncthreads();
#pragma unroll
  for (int mt = 0; mt < 2; ++mt)
#pragma unroll
    for (int nt = 0; nt < 4; ++nt) {
      int col = n0 + nt * 16 + ls;
      float bb = bV[col];
#pragma unroll
      for (int r = 0; r < 8; ++r)
        Abuf[(mt * 16 + r + hi * 8) * ASTR + col] = f2bf(acc[mt][nt][r] + bb);
    }
  __syncthreads();

  // ---- GEMM3: att = tmp @ W_o^T + b_o ; pre2 = att + liq ; LN ; state update ----
#pragma unroll
  for (int mt = 0; mt < 2; ++mt)
#pragma unroll
    for (int nt = 0; nt < 4; ++nt) acc[mt][nt] = (v8f)Z8;
  for (int kc = 0; kc < HDIM; kc += 32) {
    v16bf a0 = load_frag(&Abuf[ls * ASTR + kc + hi * 8]);
    v16bf a1 = load_frag(&Abuf[(16 + ls) * ASTR + kc + hi * 8]);
#pragma unroll
    for (int nt = 0; nt < 4; ++nt) {
      v16bf b = load_frag(wO + (size_t)(n0 + nt * 16 + ls) * HDIM + kc + hi * 8);
      acc[0][nt] = wmma_bf16(a0, b, acc[0][nt]);
      acc[1][nt] = wmma_bf16(a1, b, acc[1][nt]);
    }
  }
#pragma unroll
  for (int mt = 0; mt < 2; ++mt)
#pragma unroll
    for (int nt = 0; nt < 4; ++nt) {
      int col = n0 + nt * 16 + ls;
      float bb = bO[col];
#pragma unroll
      for (int r = 0; r < 8; ++r)
        acc[mt][nt][r] += bb + liq[mt][nt][r];
    }
#pragma unroll
  for (int mt = 0; mt < 2; ++mt)
#pragma unroll
    for (int r = 0; r < 8; ++r) {
      float s = acc[mt][0][r] + acc[mt][1][r] + acc[mt][2][r] + acc[mt][3][r];
      float q = acc[mt][0][r] * acc[mt][0][r] + acc[mt][1][r] * acc[mt][1][r] +
                acc[mt][2][r] * acc[mt][2][r] + acc[mt][3][r] * acc[mt][3][r];
#pragma unroll
      for (int msk = 1; msk < 16; msk <<= 1) {
        s += __shfl_xor(s, msk, 32);
        q += __shfl_xor(q, msk, 32);
      }
      if (ls == 0) {
        int m = mt * 16 + r + hi * 8;
        rowS[wave][m] = s;
        rowQ[wave][m] = q;
      }
    }
  __syncthreads();
  if (tid < MB2) {
    float s = 0.f, q = 0.f;
#pragma unroll
    for (int w = 0; w < 8; ++w) { s += rowS[w][tid]; q += rowQ[w][tid]; }
    float mean = s * (1.f / 512.f);
    float var = q * (1.f / 512.f) - mean * mean;
    meanA[tid] = mean;
    rstdA[tid] = rsqrtf(var + 1e-5f);
  }
  __syncthreads();
#pragma unroll
  for (int mt = 0; mt < 2; ++mt)
#pragma unroll
    for (int nt = 0; nt < 4; ++nt) {
      int col = n0 + nt * 16 + ls;
      float g = gAtt[col], bt = betaAtt[col];
      float binv = 1.f / (tau[col] + 1e-6f);
#pragma unroll
      for (int r = 0; r < 8; ++r) {
        int m = mt * 16 + r + hi * 8;
        float l2 = (acc[mt][nt][r] - meanA[m]) * rstdA[m] * g + bt;
        size_t idx = (size_t)(row0 + m) * HDIM + col;
        float hv = h[idx];
        float u = tanhf(l2 + xin[idx]);
        float hn = hv + (u - hv) * binv;
        h[idx] = hn;
        if (write_h0) h0bf[idx] = f2bf(hn);
      }
    }
}

// out = relu([h,h0] @ W1^T + b1) @ W2^T + b2
__global__ void __launch_bounds__(256) lnn_head_kernel(
    const float* __restrict__ h, const unsigned short* __restrict__ h0bf,
    const unsigned short* __restrict__ w1, const float* __restrict__ b1,
    const unsigned short* __restrict__ w2, const float* __restrict__ b2,
    float* __restrict__ out) {
  __shared__ __align__(16) unsigned short Abuf[MB * 1040];
  __shared__ __align__(16) unsigned short Ybuf[MB * ASTR];
  const int tid = threadIdx.x;
  const int row0 = blockIdx.x * MB;
  for (int i = tid; i < MB * HDIM; i += 256) {
    int r = i >> 9, c = i & 511;
    Abuf[r * 1040 + c] = f2bf(h[(size_t)(row0 + r) * HDIM + c]);
    Abuf[r * 1040 + 512 + c] = h0bf[(size_t)(row0 + r) * HDIM + c];
  }
  __syncthreads();
  const int lane = tid & 31, wave = tid >> 5;
  const int hi = lane >> 4, ls = lane & 15;
  const int n0 = wave * 64;
  v8f acc[4] = {Z8, Z8, Z8, Z8};
  for (int kc = 0; kc < 2 * HDIM; kc += 32) {
    v16bf a = load_frag(&Abuf[ls * 1040 + kc + hi * 8]);
#pragma unroll
    for (int nt = 0; nt < 4; ++nt) {
      v16bf b = load_frag(w1 + (size_t)(n0 + nt * 16 + ls) * (2 * HDIM) + kc + hi * 8);
      acc[nt] = wmma_bf16(a, b, acc[nt]);
    }
  }
#pragma unroll
  for (int nt = 0; nt < 4; ++nt) {
    int col = n0 + nt * 16 + ls;
    float bb = b1[col];
#pragma unroll
    for (int r = 0; r < 8; ++r)
      Ybuf[(r + hi * 8) * ASTR + col] = f2bf(fmaxf(acc[nt][r] + bb, 0.f));
  }
  __syncthreads();
  // second GEMM: N=128, one 16-col tile per wave
  v8f o = Z8;
  const int n0b = wave * 16;
  for (int kc = 0; kc < HDIM; kc += 32) {
    v16bf a = load_frag(&Ybuf[ls * ASTR + kc + hi * 8]);
    v16bf b = load_frag(w2 + (size_t)(n0b + ls) * HDIM + kc + hi * 8);
    o = wmma_bf16(a, b, o);
  }
  int col = n0b + ls;
  float bb = b2[col];
#pragma unroll
  for (int r = 0; r < 8; ++r)
    out[(size_t)(row0 + r + hi * 8) * OUTDIM + col] = o[r] + bb;
}

extern "C" void kernel_launch(void* const* d_in, const int* in_sizes, int n_in,
                              void* d_out, int out_size, void* d_ws, size_t ws_size,
                              hipStream_t stream) {
  const float* x        = (const float*)d_in[0];
  const float* W_in     = (const float*)d_in[1];
  const float* b_in     = (const float*)d_in[2];
  const float* W_liq    = (const float*)d_in[3];
  const float* b_liq    = (const float*)d_in[4];
  const float* g_liq    = (const float*)d_in[5];
  const float* beta_liq = (const float*)d_in[6];
  const float* W_v      = (const float*)d_in[7];
  const float* b_v      = (const float*)d_in[8];
  const float* W_o      = (const float*)d_in[9];
  const float* b_o      = (const float*)d_in[10];
  const float* g_att    = (const float*)d_in[11];
  const float* beta_att = (const float*)d_in[12];
  const float* tau      = (const float*)d_in[13];
  const float* W1       = (const float*)d_in[14];
  const float* b1       = (const float*)d_in[15];
  const float* W2       = (const float*)d_in[16];
  const float* b2       = (const float*)d_in[17];
  float* out = (float*)d_out;

  char* ws = (char*)d_ws;
  size_t off = 0;
  auto alloc = [&](size_t bytes) -> char* {
    char* p = ws + off;
    off += (bytes + 255) & ~(size_t)255;
    return p;
  };
  unsigned short* wIn_bf  = (unsigned short*)alloc((size_t)512 * 256 * 2);
  unsigned short* wLiq_bf = (unsigned short*)alloc((size_t)3 * 512 * 512 * 2);
  unsigned short* wV_bf   = (unsigned short*)alloc((size_t)512 * 512 * 2);
  unsigned short* wO_bf   = (unsigned short*)alloc((size_t)512 * 512 * 2);
  unsigned short* w1_bf   = (unsigned short*)alloc((size_t)512 * 1024 * 2);
  unsigned short* w2_bf   = (unsigned short*)alloc((size_t)128 * 512 * 2);
  float* xin              = (float*)alloc((size_t)32768 * 512 * 4);
  float* h                = (float*)alloc((size_t)32768 * 512 * 4);
  unsigned short* h0bf    = (unsigned short*)alloc((size_t)32768 * 512 * 2);

  auto cvt = [&](const float* s, unsigned short* d, int n) {
    lnn_cvt_bf16<<<(n + 255) / 256, 256, 0, stream>>>(s, d, n);
  };
  cvt(W_in,  wIn_bf,  512 * 256);
  cvt(W_liq, wLiq_bf, 3 * 512 * 512);
  cvt(W_v,   wV_bf,   512 * 512);
  cvt(W_o,   wO_bf,   512 * 512);
  cvt(W1,    w1_bf,   512 * 1024);
  cvt(W2,    w2_bf,   128 * 512);

  lnn_zero_f32<<<(32768 * 512 + 255) / 256, 256, 0, stream>>>(h, 32768 * 512);

  lnn_xin_kernel<<<NB2, 256, 0, stream>>>(x, wIn_bf, b_in, xin);

  for (int t = 0; t < 3; ++t) {  // steps == 3 per setup_inputs
    lnn_step_kernel<<<NB2, 256, 0, stream>>>(h, xin, wLiq_bf, b_liq, g_liq, beta_liq,
                                             wV_bf, b_v, wO_bf, b_o, g_att, beta_att,
                                             tau, h0bf, (t == 0) ? 1 : 0);
  }

  lnn_head_kernel<<<NB, 256, 0, stream>>>(h, h0bf, w1_bf, b1, w2_bf, b2, out);
}
